// BiLstmCRF_63359357551114
// MI455X (gfx1250) — matrix operand order repaired and checked by reading, not compiled, not run on hardware
//
#include <hip/hip_runtime.h>
#include <hip/hip_bf16.h>

// ---------------------------------------------------------------------------
// BiLSTM-CRF for MI455X (gfx1250): bf16 WMMA GEMMs, register-resident LSTM
// cell state, TDM-staged emission weights, LDS-resident CRF scans.
// ---------------------------------------------------------------------------

#define T_     256
#define B_     64
#define EMB_   300
#define EMBP   320     // EMB padded to multiple of 32 (WMMA K-step)
#define H_     256
#define G4H    1024    // 4*H
#define HID_   512
#define TAGS_  50
#define TAGSP  64

typedef __attribute__((ext_vector_type(16))) __bf16        v16bf;
typedef __attribute__((ext_vector_type(8)))  float         v8f;
typedef __attribute__((ext_vector_type(4)))  unsigned int  u32x4;
typedef __attribute__((ext_vector_type(8)))  int           i32x8;
typedef __attribute__((ext_vector_type(4)))  int           i32x4;

union FragU { v16bf bf; u32x4 q[2]; };

__device__ __forceinline__ unsigned short f2bf(float f) {
    unsigned int u = __float_as_uint(f);
    u += 0x7FFFu + ((u >> 16) & 1u);          // round-to-nearest-even
    return (unsigned short)(u >> 16);
}
__device__ __forceinline__ float bf2f(unsigned short s) {
    return __uint_as_float(((unsigned int)s) << 16);
}
__device__ __forceinline__ float sigmoidf_(float x) {
    return 1.0f / (1.0f + expf(-x));
}

// A fragment: 16x32 bf16, row-major A[M, lda]; lane = 16*hi + m holds
// K in [8*hi, 8*hi+8) and [16+8*hi, 16+8*hi+8).
__device__ __forceinline__ v16bf load_a_frag(const unsigned short* A, int lda,
                                             int m0, int k0, int lane) {
    int hi = lane >> 4, m = lane & 15;
    const unsigned short* p = A + (size_t)(m0 + m) * lda + k0;
    FragU f;
    f.q[0] = *(const u32x4*)(p + 8 * hi);
    f.q[1] = *(const u32x4*)(p + 16 + 8 * hi);
    return f.bf;
}

// B fragment: 32x16 bf16 from W stored as [N, ldb] row-major (row n == column n
// of B). Lane = 16*hi + n holds K in [8*hi, 8*hi+8) and [16+8*hi, 16+8*hi+8).
__device__ __forceinline__ v16bf load_b_frag(const unsigned short* W, int ldb,
                                             int n0, int k0, int lane) {
    int hi = lane >> 4, n = lane & 15;
    const unsigned short* p = W + (size_t)(n0 + n) * ldb + k0;
    FragU f;
    f.q[0] = *(const u32x4*)(p + 8 * hi);
    f.q[1] = *(const u32x4*)(p + 16 + 8 * hi);
    return f.bf;
}

__device__ __forceinline__ v8f wmma_bf16(v16bf a, v16bf b, v8f c) {
    return __builtin_amdgcn_wmma_f32_16x16x32_bf16(false, a, false, b,
                                                   (short)0, c, false, false);
}

// ---------------------------------------------------------------------------
// 1) Pack fp32 matrix -> bf16 with optional row/col zero padding
// ---------------------------------------------------------------------------
__global__ void pack_bf16_kernel(const float* __restrict__ in,
                                 unsigned short* __restrict__ out,
                                 int rows_in, int cols_in,
                                 int rows_out, int cols_out) {
    size_t i = (size_t)blockIdx.x * 256 + threadIdx.x;
    size_t total = (size_t)rows_out * cols_out;
    if (i >= total) return;
    int r = (int)(i / cols_out), c = (int)(i % cols_out);
    float v = (r < rows_in && c < cols_in) ? in[(size_t)r * cols_in + c] : 0.0f;
    out[i] = f2bf(v);
}

// ---------------------------------------------------------------------------
// 2) Embedding gather -> bf16, K padded to EMBP
// ---------------------------------------------------------------------------
__global__ void embed_kernel(const int* __restrict__ x,
                             const float* __restrict__ emb,
                             unsigned short* __restrict__ e_bf) {
    size_t i = (size_t)blockIdx.x * 256 + threadIdx.x;
    if (i >= (size_t)(T_ * B_) * EMBP) return;
    int row = (int)(i / EMBP), c = (int)(i % EMBP);
    float v = (c < EMB_) ? emb[(size_t)x[row] * EMB_ + c] : 0.0f;
    e_bf[i] = f2bf(v);
}

// ---------------------------------------------------------------------------
// 3) xp = e @ W_ih^T + b_ih   (M=16384, N=1024, K=320)  bf16 WMMA, fp32 out
// ---------------------------------------------------------------------------
__global__ void __launch_bounds__(256) gemm_xp_kernel(
    const unsigned short* __restrict__ A,   // [16384, 320] bf16
    const unsigned short* __restrict__ W,   // [1024, 320] bf16
    const float* __restrict__ bias,         // [1024]
    float* __restrict__ out) {              // [16384, 1024]
    int wid  = blockIdx.x * 8 + (threadIdx.x >> 5);
    int lane = threadIdx.x & 31;
    int m0 = (wid >> 4) * 16;       // 1024 M-tiles
    int n0 = (wid & 15) * 64;       // 16 N-groups of 64
    v8f acc[4] = {};
    for (int k0 = 0; k0 < EMBP; k0 += 32) {
        if (k0 + 32 < EMBP)
            __builtin_prefetch(A + (size_t)(m0 + (lane & 15)) * EMBP + k0 + 32, 0, 1);
        v16bf a = load_a_frag(A, EMBP, m0, k0, lane);
#pragma unroll
        for (int q = 0; q < 4; ++q) {
            v16bf b = load_b_frag(W, EMBP, n0 + q * 16, k0, lane);
            acc[q] = wmma_bf16(a, b, acc[q]);
        }
    }
    int hi = lane >> 4, nn = lane & 15;
#pragma unroll
    for (int q = 0; q < 4; ++q) {
        int n = n0 + q * 16 + nn;
        float bb = bias[n];
#pragma unroll
        for (int r = 0; r < 8; ++r) {
            int m = m0 + r + 8 * hi;
            out[(size_t)m * G4H + n] = acc[q][r] + bb;
        }
    }
}

// ---------------------------------------------------------------------------
// 4) LSTM recurrence: one block (512 threads = 16 waves) per direction.
//    Wave w owns unit columns u in [16w, 16w+16) for ALL FOUR gates
//    (N-tiles w, 16+w, 32+w, 48+w). i/f/g/o pre-activations for a given
//    (batch, unit) then live in the same lane/slot of the four accumulators,
//    so the cell update c = sig(f)*c + sig(i)*tanh(g), h = sig(o)*tanh(c)
//    is pure lane-local register math: c stays in VGPRs for all 256 steps,
//    LDS holds only h ([64][256] bf16, 32 KB), two barriers per step.
// ---------------------------------------------------------------------------
__global__ void __launch_bounds__(512) lstm_kernel(
    const float* __restrict__ xp_f, const float* __restrict__ xp_b,
    const unsigned short* __restrict__ whh_f,
    const unsigned short* __restrict__ whh_b,
    const float* __restrict__ bhh_f, const float* __restrict__ bhh_b,
    unsigned short* __restrict__ h_cat) {
    __shared__ unsigned short h_lds[B_ * H_];   // [64][256] bf16

    const int dir = blockIdx.x;
    const float* xp = dir ? xp_b : xp_f;
    const unsigned short* W = dir ? whh_b : whh_f;
    const float* bh = dir ? bhh_b : bhh_f;

    const int tid = threadIdx.x, lane = tid & 31, w = tid >> 5;
    const int hi = lane >> 4, nn = lane & 15;
    const int u = w * 16 + nn;                  // this lane's unit column

    for (int i = tid; i < B_ * H_; i += 512) h_lds[i] = 0;
    __syncthreads();

    const float bi = bh[u], bf_ = bh[H_ + u], bg = bh[2 * H_ + u], bo = bh[3 * H_ + u];
    float c_reg[4][8] = {};

    for (int s = 0; s < T_; ++s) {
        const int t = dir ? (T_ - 1 - s) : s;
        v8f acc[4][4] = {};
        for (int k0 = 0; k0 < H_; k0 += 32) {
            v16bf a0 = load_a_frag(h_lds, H_,  0, k0, lane);
            v16bf a1 = load_a_frag(h_lds, H_, 16, k0, lane);
            v16bf a2 = load_a_frag(h_lds, H_, 32, k0, lane);
            v16bf a3 = load_a_frag(h_lds, H_, 48, k0, lane);
#pragma unroll
            for (int g = 0; g < 4; ++g) {
                v16bf b = load_b_frag(W, H_, g * H_ + w * 16, k0, lane);
                acc[g][0] = wmma_bf16(a0, b, acc[g][0]);
                acc[g][1] = wmma_bf16(a1, b, acc[g][1]);
                acc[g][2] = wmma_bf16(a2, b, acc[g][2]);
                acc[g][3] = wmma_bf16(a3, b, acc[g][3]);
            }
        }
        // lane-local cell update (no LDS): gates = acc + xp[t] + b_hh
        const float* xpt = xp + (size_t)t * B_ * G4H;
        unsigned short hb[4][8];
#pragma unroll
        for (int mt = 0; mt < 4; ++mt) {
#pragma unroll
            for (int r = 0; r < 8; ++r) {
                int b = mt * 16 + r + 8 * hi;
                const float* xr = xpt + (size_t)b * G4H + u;
                float vi = sigmoidf_(acc[0][mt][r] + xr[0]        + bi);
                float vf = sigmoidf_(acc[1][mt][r] + xr[H_]       + bf_);
                float vg = tanhf   (acc[2][mt][r] + xr[2 * H_]    + bg);
                float vo = sigmoidf_(acc[3][mt][r] + xr[3 * H_]   + bo);
                float cc = vf * c_reg[mt][r] + vi * vg;
                c_reg[mt][r] = cc;
                hb[mt][r] = f2bf(vo * tanhf(cc));
            }
        }
        __syncthreads();   // all waves done reading h_lds for this step
#pragma unroll
        for (int mt = 0; mt < 4; ++mt) {
#pragma unroll
            for (int r = 0; r < 8; ++r) {
                int b = mt * 16 + r + 8 * hi;
                h_lds[b * H_ + u] = hb[mt][r];
                h_cat[((size_t)t * B_ + b) * HID_ + dir * H_ + u] = hb[mt][r];
            }
        }
        __syncthreads();   // h visible before next step's A-fragment reads
    }
}

// ---------------------------------------------------------------------------
// 5) emissions = h_cat @ W_out^T + b_out  (M=16384, N=64(pad 50), K=512)
//    B matrix (64x512 bf16, 64 KB) staged into LDS once per block via the
//    Tensor Data Mover; B fragments then come from LDS (ds_load_b128).
// ---------------------------------------------------------------------------
__global__ void __launch_bounds__(256) gemm_em_kernel(
    const unsigned short* __restrict__ Hc,   // [16384, 512] bf16
    const unsigned short* __restrict__ Wo,   // [64, 512] bf16 (rows 50..63 = 0)
    const float* __restrict__ b_out,         // [50]
    float* __restrict__ em) {                // [16384, 50]
    extern __shared__ char smem_em[];
    unsigned short* wo_lds = (unsigned short*)smem_em;  // [64][512] bf16

    if (threadIdx.x < 32) {
        // Tensor DMA descriptor: 2D bf16 tensor, 512 x 64, tile = whole tensor.
        unsigned long long ga = (unsigned long long)Wo;
        u32x4 g0;
        g0[0] = 1u;                                   // count=1 (valid), user mode
        g0[1] = 0u;                                   // lds_addr = base of dyn LDS
        g0[2] = (unsigned int)(ga & 0xFFFFFFFFu);     // global_addr[31:0]
        g0[3] = (unsigned int)((ga >> 32) & 0x1FFFFFFu) | (2u << 30);  // addr hi | type=2
        i32x8 g1;
        g1[0] = (int)(1u << 16);                      // wg_mask=0 | data_size=1 (2B)
        g1[1] = (int)((512u & 0xFFFFu) << 16);        // tensor_dim0 = 512 (lo16)
        g1[2] = (int)(64u << 16);                     // dim0 hi=0 | tensor_dim1 = 64
        g1[3] = (int)(512u << 16);                    // dim1 hi=0 | tile_dim0 = 512
        g1[4] = (int)64u;                             // tile_dim1 = 64 | tile_dim2 = 0
        g1[5] = (int)512u;                            // tensor_dim0_stride = 512
        g1[6] = 0;
        g1[7] = 0;
        i32x4 gz = {0, 0, 0, 0};
#if __clang_major__ >= 23
        i32x8 gz8 = {0, 0, 0, 0, 0, 0, 0, 0};
        __builtin_amdgcn_tensor_load_to_lds(g0, g1, gz, gz, gz8, 0);
#else
        __builtin_amdgcn_tensor_load_to_lds(g0, g1, gz, gz, 0);
#endif
        __builtin_amdgcn_s_wait_tensorcnt(0);
    }
    __syncthreads();

    int wid  = blockIdx.x * 8 + (threadIdx.x >> 5);  // m-tile 0..1023
    int lane = threadIdx.x & 31;
    int m0 = wid * 16;
    v8f acc[4] = {};
    for (int k0 = 0; k0 < HID_; k0 += 32) {
        v16bf a = load_a_frag(Hc, HID_, m0, k0, lane);
#pragma unroll
        for (int q = 0; q < 4; ++q) {
            v16bf b = load_b_frag(wo_lds, HID_, q * 16, k0, lane);
            acc[q] = wmma_bf16(a, b, acc[q]);
        }
    }
    int hi = lane >> 4, nn = lane & 15;
#pragma unroll
    for (int q = 0; q < 4; ++q) {
        int n = q * 16 + nn;
        if (n < TAGS_) {
            float bb = b_out[n];
#pragma unroll
            for (int r = 0; r < 8; ++r) {
                int m = m0 + r + 8 * hi;
                em[(size_t)m * TAGS_ + n] = acc[q][r] + bb;
            }
        }
    }
}

// ---------------------------------------------------------------------------
// 6) CRF loss: gold-path score + forward algorithm (log-sum-exp DP) in LDS
// ---------------------------------------------------------------------------
__global__ void __launch_bounds__(1024) crf_loss_kernel(
    const float* __restrict__ em, const int* __restrict__ y,
    const unsigned char* __restrict__ mask,
    const float* __restrict__ start, const float* __restrict__ endt,
    const float* __restrict__ trans, float* __restrict__ out_loss) {
    __shared__ float sc[B_][TAGS_];
    __shared__ float ns[B_][TAGS_];
    __shared__ float tr[TAGS_][TAGS_];
    __shared__ float num[B_];
    __shared__ float lsum[B_];
    int tid = threadIdx.x;
    for (int i = tid; i < TAGS_ * TAGS_; i += 1024) tr[i / TAGS_][i % TAGS_] = trans[i];
    for (int i = tid; i < B_ * TAGS_; i += 1024) {
        int b = i / TAGS_, j = i % TAGS_;
        sc[b][j] = start[j] + em[(size_t)b * TAGS_ + j];
    }
    if (tid < B_) {  // numerator: gold path score
        int b = tid;
        int prev = y[b];
        float nm = start[prev] + em[(size_t)b * TAGS_ + prev];
        int cnt = mask[b] ? 1 : 0;
        for (int t = 1; t < T_; ++t) {
            int yt = y[t * B_ + b];
            if (mask[t * B_ + b]) {
                nm += trans[prev * TAGS_ + yt] + em[((size_t)t * B_ + b) * TAGS_ + yt];
                ++cnt;
            }
            prev = yt;
        }
        nm += endt[y[(cnt - 1) * B_ + b]];
        num[b] = nm;
    }
    __syncthreads();
    for (int t = 1; t < T_; ++t) {
        const float* emt = em + (size_t)t * B_ * TAGS_;
        for (int i = tid; i < B_ * TAGS_; i += 1024) {
            int b = i / TAGS_, j = i % TAGS_;
            float mx = -3.4e38f;
            for (int k = 0; k < TAGS_; ++k) mx = fmaxf(mx, sc[b][k] + tr[k][j]);
            float s = 0.0f;
            for (int k = 0; k < TAGS_; ++k) s += expf(sc[b][k] + tr[k][j] - mx);
            float v = mx + logf(s) + emt[(size_t)b * TAGS_ + j];
            ns[b][j] = mask[t * B_ + b] ? v : sc[b][j];
        }
        __syncthreads();
        for (int i = tid; i < B_ * TAGS_; i += 1024) sc[i / TAGS_][i % TAGS_] = ns[i / TAGS_][i % TAGS_];
        __syncthreads();
    }
    if (tid < B_) {
        int b = tid;
        float mx = -3.4e38f;
        for (int j = 0; j < TAGS_; ++j) mx = fmaxf(mx, sc[b][j] + endt[j]);
        float s = 0.0f;
        for (int j = 0; j < TAGS_; ++j) s += expf(sc[b][j] + endt[j] - mx);
        lsum[b] = (mx + logf(s)) - num[b];
    }
    __syncthreads();
    if (tid == 0) {
        float tot = 0.0f;
        for (int b = 0; b < B_; ++b) tot += lsum[b];
        out_loss[0] = tot;
    }
}

// ---------------------------------------------------------------------------
// 7) Viterbi decode: max-DP with history + backtrack; pred as float tags
// ---------------------------------------------------------------------------
__global__ void __launch_bounds__(1024) viterbi_kernel(
    const float* __restrict__ em, const unsigned char* __restrict__ mask,
    const float* __restrict__ start, const float* __restrict__ endt,
    const float* __restrict__ trans, int* __restrict__ hist,
    float* __restrict__ pred_out) {
    __shared__ float sc[B_][TAGS_];
    __shared__ float ns[B_][TAGS_];
    __shared__ float tr[TAGS_][TAGS_];
    int tid = threadIdx.x;
    for (int i = tid; i < TAGS_ * TAGS_; i += 1024) tr[i / TAGS_][i % TAGS_] = trans[i];
    for (int i = tid; i < B_ * TAGS_; i += 1024) {
        int b = i / TAGS_, j = i % TAGS_;
        sc[b][j] = start[j] + em[(size_t)b * TAGS_ + j];
    }
    __syncthreads();
    for (int t = 1; t < T_; ++t) {
        const float* emt = em + (size_t)t * B_ * TAGS_;
        for (int i = tid; i < B_ * TAGS_; i += 1024) {
            int b = i / TAGS_, j = i % TAGS_;
            float mx = sc[b][0] + tr[0][j];
            int am = 0;
            for (int k = 1; k < TAGS_; ++k) {
                float v = sc[b][k] + tr[k][j];
                if (v > mx) { mx = v; am = k; }
            }
            hist[(((size_t)(t - 1)) * B_ + b) * TAGS_ + j] = am;
            float v = mx + emt[(size_t)b * TAGS_ + j];
            ns[b][j] = mask[t * B_ + b] ? v : sc[b][j];
        }
        __syncthreads();
        for (int i = tid; i < B_ * TAGS_; i += 1024) sc[i / TAGS_][i % TAGS_] = ns[i / TAGS_][i % TAGS_];
        __syncthreads();
    }
    if (tid < B_) {
        int b = tid;
        float mx = sc[b][0] + endt[0];
        int cur = 0;
        for (int j = 1; j < TAGS_; ++j) {
            float v = sc[b][j] + endt[j];
            if (v > mx) { mx = v; cur = j; }
        }
        for (int t = T_ - 1; t >= 1; --t) {
            pred_out[(size_t)t * B_ + b] = (float)cur;
            int prev = hist[(((size_t)(t - 1)) * B_ + b) * TAGS_ + cur];
            cur = mask[t * B_ + b] ? prev : cur;
        }
        pred_out[b] = (float)cur;
    }
}

// ---------------------------------------------------------------------------
extern "C" void kernel_launch(void* const* d_in, const int* in_sizes, int n_in,
                              void* d_out, int out_size, void* d_ws, size_t ws_size,
                              hipStream_t stream) {
    const int*   x       = (const int*)d_in[0];
    const int*   y       = (const int*)d_in[1];
    const unsigned char* mask = (const unsigned char*)d_in[2];
    const float* emb     = (const float*)d_in[3];
    const float* W_ih_f  = (const float*)d_in[4];
    const float* W_hh_f  = (const float*)d_in[5];
    const float* b_ih_f  = (const float*)d_in[6];
    const float* b_hh_f  = (const float*)d_in[7];
    const float* W_ih_b  = (const float*)d_in[8];
    const float* W_hh_b  = (const float*)d_in[9];
    const float* b_ih_b  = (const float*)d_in[10];
    const float* b_hh_b  = (const float*)d_in[11];
    const float* W_out   = (const float*)d_in[12];
    const float* b_out   = (const float*)d_in[13];
    const float* start_t = (const float*)d_in[14];
    const float* end_t   = (const float*)d_in[15];
    const float* trans   = (const float*)d_in[16];
    float* out = (float*)d_out;

    // workspace carve-up (all region sizes are multiples of 256B)
    char* w = (char*)d_ws;
    unsigned short* e_bf   = (unsigned short*)w; w += (size_t)T_ * B_ * EMBP * 2;
    unsigned short* wih_f  = (unsigned short*)w; w += (size_t)G4H * EMBP * 2;
    unsigned short* wih_b  = (unsigned short*)w; w += (size_t)G4H * EMBP * 2;
    unsigned short* whh_f  = (unsigned short*)w; w += (size_t)G4H * H_ * 2;
    unsigned short* whh_b  = (unsigned short*)w; w += (size_t)G4H * H_ * 2;
    unsigned short* wout   = (unsigned short*)w; w += (size_t)TAGSP * HID_ * 2;
    float* xp_f            = (float*)w;          w += (size_t)T_ * B_ * G4H * 4;
    float* xp_b            = (float*)w;          w += (size_t)T_ * B_ * G4H * 4;
    unsigned short* h_cat  = (unsigned short*)w; w += (size_t)T_ * B_ * HID_ * 2;
    float* em              = (float*)w;          w += (size_t)T_ * B_ * TAGS_ * 4;
    int* hist              = (int*)w;            w += (size_t)(T_ - 1) * B_ * TAGS_ * 4;

    // 1) weight packing (fp32 -> bf16, pad K/N)
    {
        size_t n1 = (size_t)G4H * EMBP;
        pack_bf16_kernel<<<dim3((n1 + 255) / 256), dim3(256), 0, stream>>>(
            W_ih_f, wih_f, G4H, EMB_, G4H, EMBP);
        pack_bf16_kernel<<<dim3((n1 + 255) / 256), dim3(256), 0, stream>>>(
            W_ih_b, wih_b, G4H, EMB_, G4H, EMBP);
        size_t n2 = (size_t)G4H * H_;
        pack_bf16_kernel<<<dim3((n2 + 255) / 256), dim3(256), 0, stream>>>(
            W_hh_f, whh_f, G4H, H_, G4H, H_);
        pack_bf16_kernel<<<dim3((n2 + 255) / 256), dim3(256), 0, stream>>>(
            W_hh_b, whh_b, G4H, H_, G4H, H_);
        size_t n3 = (size_t)TAGSP * HID_;
        pack_bf16_kernel<<<dim3((n3 + 255) / 256), dim3(256), 0, stream>>>(
            W_out, wout, TAGS_, HID_, TAGSP, HID_);
    }
    // 2) embedding gather -> bf16
    {
        size_t n = (size_t)T_ * B_ * EMBP;
        embed_kernel<<<dim3((n + 255) / 256), dim3(256), 0, stream>>>(x, emb, e_bf);
    }
    // 3) input projections (WMMA GEMM), b_ih fused
    gemm_xp_kernel<<<dim3(2048), dim3(256), 0, stream>>>(e_bf, wih_f, b_ih_f, xp_f);
    gemm_xp_kernel<<<dim3(2048), dim3(256), 0, stream>>>(e_bf, wih_b, b_ih_b, xp_b);
    // 4) both LSTM directions: register-resident cell state, 32 KB LDS h buffer
    lstm_kernel<<<dim3(2), dim3(512), 0, stream>>>(
        xp_f, xp_b, whh_f, whh_b, b_hh_f, b_hh_b, h_cat);
    // 5) emission projection (WMMA GEMM, TDM-staged B panel in 64 KB dyn LDS)
    gemm_em_kernel<<<dim3(128), dim3(256), 65536, stream>>>(h_cat, wout, b_out, em);
    // 6) CRF loss -> out[0]
    crf_loss_kernel<<<dim3(1), dim3(1024), 0, stream>>>(
        em, y, mask, start_t, end_t, trans, out);
    // 7) Viterbi decode -> out[1..16384] (tags as float)
    viterbi_kernel<<<dim3(1), dim3(1024), 0, stream>>>(
        em, mask, start_t, end_t, trans, hist, out + 1);
}
